// SpikingTradingAgent_72301479460916
// MI455X (gfx1250) — compile-verified, matrix operand order checked by reading, not run on hardware
//
#include <hip/hip_runtime.h>

typedef __attribute__((ext_vector_type(16))) _Float16 v16h;
typedef __attribute__((ext_vector_type(8)))  _Float16 v8h;
typedef __attribute__((ext_vector_type(8)))  float    v8f;
typedef __attribute__((ext_vector_type(4)))  unsigned uv4;
typedef __attribute__((ext_vector_type(4)))  int      iv4;
typedef __attribute__((ext_vector_type(8)))  int      iv8;

#define B_  4096
#define I_  512
#define H_  1024
#define P_  4
#define T_  32
#define O_  3

// ---------------------------------------------------------------------------
// f32 -> f16 elementwise convert
// ---------------------------------------------------------------------------
__global__ void cvt_f32_to_f16_kernel(const float* __restrict__ src,
                                      _Float16* __restrict__ dst, int n) {
  int i = blockIdx.x * blockDim.x + threadIdx.x;
  if (i < n) dst[i] = (_Float16)src[i];
}

// ---------------------------------------------------------------------------
// Pack a row-major f32 [K x N] matrix (batched) into WMMA B-fragment order:
//   fragment id = nt*(K/32)+kc, within fragment: lane*16 + j halves,
//   lane<16 -> N=nt*16+lane, K0=kc*32 ;  lane>=16 -> K0=kc*32+8
//   half j<8 -> K=K0+j ; j>=8 -> K=K0+16+(j-8)
// A wave then loads its v16h B fragment as one contiguous 32B chunk per lane.
// ---------------------------------------------------------------------------
__global__ void pack_b_frag_kernel(const float* __restrict__ W,
                                   _Float16* __restrict__ out, int K, int N) {
  int idx = blockIdx.x * blockDim.x + threadIdx.x;
  int total = K * N;
  if (idx >= total) return;
  int batch = blockIdx.y;
  const float* Wb = W + (size_t)batch * K * N;
  _Float16* ob = out + (size_t)batch * K * N;
  int fragId = idx >> 9;
  int within = idx & 511;
  int lane = within >> 4;
  int j = within & 15;
  int KC = K >> 5;
  int kc = fragId % KC;
  int nt = fragId / KC;
  int n = nt * 16 + (lane & 15);
  int k = kc * 32 + ((lane < 16) ? 0 : 8) + ((j < 8) ? j : j + 8);
  ob[idx] = (_Float16)Wb[(size_t)k * N + n];
}

// ---------------------------------------------------------------------------
// Generic WMMA f16 GEMM: C[M,N] = act(A[M,K] * Bpacked + bias)
// One wave per 16x16 C tile. ACT: 0 none, 1 sigmoid, 2 relu.
// ---------------------------------------------------------------------------
template <int ACT, int HALF_OUT>
__global__ __launch_bounds__(256)
void gemm_wmma_kernel(const _Float16* __restrict__ A,
                      const _Float16* __restrict__ Bp,
                      const float* __restrict__ bias,
                      float* __restrict__ Cf,
                      _Float16* __restrict__ Ch,
                      int M, int N, int K) {
  int wave = (blockIdx.x * blockDim.x + threadIdx.x) >> 5;
  int lane = threadIdx.x & 31;
  int Mt = M >> 4;
  int mt = wave % Mt;
  int nt = wave / Mt;
  int KC = K >> 5;

  v8f c = {};
  int row = mt * 16 + (lane & 15);
  int k0l = (lane < 16) ? 0 : 8;
  const _Float16* arow = A + (size_t)row * K + k0l;
  const _Float16* bfrag = Bp + (size_t)nt * KC * 512 + lane * 16;

  for (int kc = 0; kc < KC; ++kc) {
    v8h alo = *(const v8h*)(arow + kc * 32);
    v8h ahi = *(const v8h*)(arow + kc * 32 + 16);
    v16h a = __builtin_shufflevector(alo, ahi, 0, 1, 2, 3, 4, 5, 6, 7,
                                     8, 9, 10, 11, 12, 13, 14, 15);
    v16h b = *(const v16h*)(bfrag + (size_t)kc * 512);
    c = __builtin_amdgcn_wmma_f32_16x16x32_f16(false, a, false, b,
                                               (short)0, c, false, false);
  }

  int col = nt * 16 + (lane & 15);
  float bv = bias[col];
  int rbase = mt * 16 + ((lane < 16) ? 0 : 8);
#pragma unroll
  for (int r = 0; r < 8; ++r) {
    float v = c[r] + bv;
    if (ACT == 1) v = 1.0f / (1.0f + __expf(-v));
    if (ACT == 2) v = fmaxf(v, 0.0f);
    size_t o = (size_t)(rbase + r) * N + col;
    if (HALF_OUT) Ch[o] = (_Float16)v;
    else          Cf[o] = v;
  }
}

// ---------------------------------------------------------------------------
// Bernoulli spike train, bit-packed: bits[t][b][w] (32 input channels / word)
// ---------------------------------------------------------------------------
__global__ void spike_bits_kernel(const float* __restrict__ noise,
                                  const float* __restrict__ enc,
                                  unsigned* __restrict__ bits) {
  int id = blockIdx.x * blockDim.x + threadIdx.x;  // B*T*16 threads
  if (id >= B_ * T_ * 16) return;
  int w = id & 15;
  int t = (id >> 4) & 31;
  int b = id >> 9;
  const float* np = noise + ((size_t)b * T_ + t) * I_ + w * 32;
  const float* ep = enc + (size_t)b * I_ + w * 32;
  unsigned m = 0;
#pragma unroll
  for (int j = 0; j < 32; ++j) m |= (np[j] < ep[j]) ? (1u << j) : 0u;
  bits[((size_t)t * B_ + b) * 16 + w] = m;
}

// ---------------------------------------------------------------------------
// Pathway softmax: pw[b][p] = softmax(x @ sel_W + sel_b)
// ---------------------------------------------------------------------------
__global__ void pathway_kernel(const float* __restrict__ x,
                               const float* __restrict__ selW,
                               const float* __restrict__ selb,
                               float* __restrict__ pw) {
  int b = blockIdx.x * blockDim.x + threadIdx.x;
  if (b >= B_) return;
  float acc[P_] = {selb[0], selb[1], selb[2], selb[3]};
  const float* xr = x + (size_t)b * I_;
  for (int i = 0; i < I_; ++i) {
    float xv = xr[i];
#pragma unroll
    for (int p = 0; p < P_; ++p) acc[p] += xv * selW[i * P_ + p];
  }
  float mx = fmaxf(fmaxf(acc[0], acc[1]), fmaxf(acc[2], acc[3]));
  float e[P_], s = 0.f;
#pragma unroll
  for (int p = 0; p < P_; ++p) { e[p] = __expf(acc[p] - mx); s += e[p]; }
  float inv = 1.0f / s;
#pragma unroll
  for (int p = 0; p < P_; ++p) pw[(size_t)b * P_ + p] = e[p] * inv;
}

// ---------------------------------------------------------------------------
// LIF recurrent kernel.
// Block = 256 threads = 8 waves = (2 batch sub-tiles) x (4 pathways).
// Each wave: one pathway p, one 16-row batch sub-tile, two 16x16 H tiles.
// State (mem/ref/acc) lives in C-fragment VGPRs across all T steps;
// the compiler register-hoists the 32 B fragments out of the t-loop.
//
// CDNA5 data movers used:
//  * TDM: one tensor_load_to_lds moves the whole 128KB lif_W slice
//    (2-D tile: 4 pathway-rows x 32KB, row stride 1MB) -> TENSORcnt.
//  * Async VMEM->LDS: double-buffered ping-pong prefetch of each step's
//    2KB spike-bit rows via global_load_async_to_lds_b128 -> ASYNCcnt.
//  * WMMA f16: A fragments synthesized from bit-packed spikes with a
//    3-op/dword bit-spread + v_mul_u32_u24 trick (co-executes with XDL).
// Final sum over pathways via LDS reduction.
// ---------------------------------------------------------------------------
__global__ __launch_bounds__(256, 1)
void lif_kernel(const unsigned* __restrict__ bits,
                const _Float16* __restrict__ Whp,
                const float* __restrict__ lif_b,
                const float* __restrict__ pw,
                float* __restrict__ msr) {
  extern __shared__ char smem[];
  _Float16* sW = (_Float16*)smem;               // [4p][2nt][16kc][512] = 128KB
  float* sRed = (float*)(smem + 131072);        // [2mt][4p][2j][8r][32] = 16KB
  unsigned* sBits = (unsigned*)(smem + 147456); // 2 x 32rows x 20w (pad) = 5KB

  int tid = threadIdx.x;
  int lane = tid & 31;
  int wave = tid >> 5;
  int mt = wave >> 2;                          // 0..1 batch sub-tile
  int p = wave & 3;                            // pathway id
  int bBase = blockIdx.x * 32;
  int hBase = blockIdx.y * 32;

  // ---- Stage W slice into LDS with one Tensor Data Mover descriptor ----
  if (wave == 0) {
    unsigned long long ga = (unsigned long long)(const void*)
        (Whp + (size_t)blockIdx.y * (2 * 16 * 512));
    unsigned ldsa = (unsigned)(size_t)(void*)sW;
    uv4 g0;
    g0[0] = 1u;                                           // count=1 (valid D#)
    g0[1] = ldsa;                                         // lds_addr
    g0[2] = (unsigned)ga;                                 // global_addr lo
    g0[3] = (unsigned)((ga >> 32) & 0x01FFFFFFu) | 0x80000000u; // hi | type=2
    iv8 g1;
    g1[0] = (int)(1u << 16);                  // data_size=1 -> 2-byte elems
    g1[1] = (int)(16384u << 16);              // tensor_dim0 = 16384 (lo16<<16)
    g1[2] = (int)((16384u >> 16) | (4u << 16)); // dim0 hi | tensor_dim1=4
    g1[3] = (int)(16384u << 16);              // dim1 hi(0) | tile_dim0=16384
    g1[4] = 4;                                // tile_dim1 = 4 rows
    g1[5] = (int)(I_ * H_);                   // tensor_dim0_stride = 524288
    g1[6] = 0;
    g1[7] = 0;
    iv4 gz = {0, 0, 0, 0};
#if __clang_major__ >= 23
    iv8 gz8 = {0, 0, 0, 0, 0, 0, 0, 0};
    __builtin_amdgcn_tensor_load_to_lds(g0, g1, gz, gz, gz8, 0);
#else
    __builtin_amdgcn_tensor_load_to_lds(g0, g1, gz, gz, 0);
#endif
    __builtin_amdgcn_s_wait_tensorcnt(0);
  }

  // ---- Async ping-pong prefetch of spike-bit rows (2KB/step) ----
  unsigned sBitsBase = (unsigned)(size_t)(void*)sBits;
  auto issueBits = [&](int tt) {
    if (tid < 128) {  // 128 threads x 16B = 2KB (rows bBase..bBase+31)
      unsigned la = sBitsBase + (unsigned)((tt & 1) * 2560 +
                                           (tid >> 2) * 80 + (tid & 3) * 16);
      unsigned long long ga = (unsigned long long)(const void*)
          (bits + ((size_t)tt * B_ + bBase) * 16) +
          (unsigned long long)(tid * 16);
      asm volatile("global_load_async_to_lds_b128 %0, %1, off"
                   :: "v"(la), "v"(ga) : "memory");
    }
  };
  issueBits(0);

  int hiSel = (lane < 16) ? 0 : 8;

  float pwr[8];
#pragma unroll
  for (int r = 0; r < 8; ++r)
    pwr[r] = pw[(size_t)(bBase + mt * 16 + hiSel + r) * P_ + p];

  float biasv[2];
#pragma unroll
  for (int j = 0; j < 2; ++j)
    biasv[j] = lif_b[(size_t)p * H_ + hBase + j * 16 + (lane & 15)];

  v8f mem[2], ref[2], acc[2];
  {
    v8f z = {};
    mem[0] = z; mem[1] = z; ref[0] = z; ref[1] = z; acc[0] = z; acc[1] = z;
  }

  // per-lane base of this wave's B fragments in LDS (j stride = 16*512 halves)
  const _Float16* sWp = sW + (size_t)(p * 2) * 16 * 512 + lane * 16;

  for (int t = 0; t < T_; ++t) {
    // bits for step t (and the TDM W data at t==0) are ready after this
    asm volatile("s_wait_asynccnt 0x0" ::: "memory");
    __syncthreads();
    if (t + 1 < T_) issueBits(t + 1);

    // read this row's 16 spike bit-words from LDS (conflict-free padded)
    const unsigned* browL = sBits + (t & 1) * 640 + (mt * 16 + (lane & 15)) * 20;
    unsigned wv[16];
    *(uint4*)(&wv[0])  = *(const uint4*)(browL + 0);
    *(uint4*)(&wv[4])  = *(const uint4*)(browL + 4);
    *(uint4*)(&wv[8])  = *(const uint4*)(browL + 8);
    *(uint4*)(&wv[12]) = *(const uint4*)(browL + 12);

    v8f cur[2];
#pragma unroll
    for (int j = 0; j < 2; ++j) {
      v8f cv;
#pragma unroll
      for (int r = 0; r < 8; ++r) cv[r] = biasv[j];
      cur[j] = cv;
    }

    // software-pipelined B fragments (compiler register-hoists these)
    v16h b0 = *(const v16h*)(sWp + 0);
    v16h b1 = *(const v16h*)(sWp + 16 * 512);

#pragma unroll
    for (int kc = 0; kc < 16; ++kc) {
      int kn = (kc + 1) & 15;
      v16h b0n = *(const v16h*)(sWp + (size_t)kn * 512);
      v16h b1n = *(const v16h*)(sWp + (size_t)(16 + kn) * 512);

      // A fragment: 16 bits -> 16 packed f16 {0,1} in 8 dwords,
      // 3 full-rate VALU per dword via bit-spread + mul_u24.
      unsigned wb = wv[kc];
      unsigned lo = (wb >> hiSel) & 0xFFu;
      unsigned hi = (wb >> (hiSel + 16)) & 0xFFu;
      unsigned zlo = lo | (lo << 15);
      unsigned zhi = hi | (hi << 15);
      union { v16h h; unsigned d[8]; } au;
#pragma unroll
      for (int dd = 0; dd < 4; ++dd) {
        au.d[dd]     = __umul24((zlo >> (2 * dd)) & 0x10001u, 0x3C00u);
        au.d[dd + 4] = __umul24((zhi >> (2 * dd)) & 0x10001u, 0x3C00u);
      }

      cur[0] = __builtin_amdgcn_wmma_f32_16x16x32_f16(
          false, au.h, false, b0, (short)0, cur[0], false, false);
      cur[1] = __builtin_amdgcn_wmma_f32_16x16x32_f16(
          false, au.h, false, b1, (short)0, cur[1], false, false);
      b0 = b0n;
      b1 = b1n;
    }

    // LIF dynamics + pathway-weighted spike accumulation (own p only)
#pragma unroll
    for (int j = 0; j < 2; ++j) {
#pragma unroll
      for (int r = 0; r < 8; ++r) {
        float m = 0.9f * mem[j][r] + cur[j][r];
        float rf = ref[j][r];
        float sp = (m >= 1.0f && rf == 0.0f) ? 1.0f : 0.0f;
        m -= m * sp;                                // mem *= (1 - spike)
        rf = fmaxf(rf - 1.0f + sp * 2.0f, 0.0f);    // refractory update
        mem[j][r] = m;
        ref[j][r] = rf;
        acc[j][r] += sp * pwr[r];
      }
    }
  }

  // ---- Reduce partial accumulators over the 4 pathway-waves via LDS ----
#pragma unroll
  for (int j = 0; j < 2; ++j)
#pragma unroll
    for (int r = 0; r < 8; ++r)
      sRed[(((mt * 4 + p) * 2 + j) * 8 + r) * 32 + lane] = acc[j][r];
  __syncthreads();

  const float invT = 1.0f / (float)T_;
  for (int e = tid; e < 1024; e += 256) {
    int lane2 = e & 31;
    int r = (e >> 5) & 7;
    int j = (e >> 8) & 1;
    int mt2 = e >> 9;
    float s = 0.f;
#pragma unroll
    for (int pp = 0; pp < 4; ++pp)
      s += sRed[(((mt2 * 4 + pp) * 2 + j) * 8 + r) * 32 + lane2];
    int row = bBase + mt2 * 16 + ((lane2 < 16) ? 0 : 8) + r;
    int col = hBase + j * 16 + (lane2 & 15);
    msr[(size_t)row * H_ + col] = s * invT;
  }
}

// ---------------------------------------------------------------------------
// Final tiny head: out[b][o] = hid[b][:] @ out_W2 + out_b2  (N=3)
// ---------------------------------------------------------------------------
__global__ void head2_kernel(const _Float16* __restrict__ hid,
                             const float* __restrict__ W2,
                             const float* __restrict__ b2,
                             float* __restrict__ out) {
  int b = blockIdx.x * blockDim.x + threadIdx.x;
  if (b >= B_) return;
  float a0 = b2[0], a1 = b2[1], a2 = b2[2];
  const _Float16* hr = hid + (size_t)b * (H_ / 2);
  for (int j = 0; j < H_ / 2; ++j) {
    float h = (float)hr[j];
    a0 += h * W2[j * 3 + 0];
    a1 += h * W2[j * 3 + 1];
    a2 += h * W2[j * 3 + 2];
  }
  out[b * 3 + 0] = a0;
  out[b * 3 + 1] = a1;
  out[b * 3 + 2] = a2;
}

// ---------------------------------------------------------------------------
extern "C" void kernel_launch(void* const* d_in, const int* in_sizes, int n_in,
                              void* d_out, int out_size, void* d_ws, size_t ws_size,
                              hipStream_t stream) {
  const float* x      = (const float*)d_in[0];
  const float* noise  = (const float*)d_in[1];
  /* d_in[2] = num_steps (fixed T=32) */
  const float* enc_W  = (const float*)d_in[3];
  const float* enc_b  = (const float*)d_in[4];
  const float* sel_W  = (const float*)d_in[5];
  const float* sel_b  = (const float*)d_in[6];
  const float* lif_W  = (const float*)d_in[7];
  const float* lif_b  = (const float*)d_in[8];
  const float* out_W1 = (const float*)d_in[9];
  const float* out_b1 = (const float*)d_in[10];
  const float* out_W2 = (const float*)d_in[11];
  const float* out_b2 = (const float*)d_in[12];

  float* out     = (float*)d_out;
  float* out_pw  = out + (size_t)B_ * O_;
  float* out_msr = out + (size_t)B_ * O_ + (size_t)B_ * P_;

  char* ws = (char*)d_ws;
  float*    encoded = (float*)(ws);                       // 8 MB (reused -> msrh)
  unsigned* bits    = (unsigned*)(ws + (8u  << 20));      // 8 MB
  _Float16* Whp     = (_Float16*)(ws + (16u << 20));      // 4 MB
  _Float16* BpW1    = (_Float16*)(ws + (20u << 20));      // 1 MB
  _Float16* BpEnc   = (_Float16*)(ws + (21u << 20));      // 0.5 MB
  _Float16* xh      = (_Float16*)(ws + (22u << 20));      // 4 MB (reused -> hidh)
  _Float16* msrh    = (_Float16*)(ws);                    // reuse after spikes
  _Float16* hidh    = (_Float16*)(ws + (22u << 20));      // reuse after encoder

  // Pack B matrices into WMMA fragment order
  pack_b_frag_kernel<<<dim3((I_ * I_ + 255) / 256, 1), 256, 0, stream>>>(
      enc_W, BpEnc, I_, I_);
  pack_b_frag_kernel<<<dim3((H_ * (H_ / 2) + 255) / 256, 1), 256, 0, stream>>>(
      out_W1, BpW1, H_, H_ / 2);
  pack_b_frag_kernel<<<dim3((I_ * H_ + 255) / 256, P_), 256, 0, stream>>>(
      lif_W, Whp, I_, H_);

  // x -> f16
  cvt_f32_to_f16_kernel<<<(B_ * I_ + 255) / 256, 256, 0, stream>>>(
      x, xh, B_ * I_);

  // encoded = sigmoid(x @ enc_W + enc_b)
  {
    int tiles = (B_ / 16) * (I_ / 16);
    gemm_wmma_kernel<1, 0><<<tiles / 8, 256, 0, stream>>>(
        xh, BpEnc, enc_b, encoded, nullptr, B_, I_, I_);
  }

  // spike bits
  spike_bits_kernel<<<(B_ * T_ * 16 + 255) / 256, 256, 0, stream>>>(
      noise, encoded, bits);

  // pathway softmax -> output region 2
  pathway_kernel<<<(B_ + 127) / 128, 128, 0, stream>>>(
      x, sel_W, sel_b, out_pw);

  // LIF recurrence -> mean spike rate (output region 3)
  lif_kernel<<<dim3(B_ / 32, H_ / 32), 256, 152576, stream>>>(
      bits, Whp, lif_b, out_pw, out_msr);

  // head: msr -> f16, WMMA GEMM + relu, tiny final GEMM
  cvt_f32_to_f16_kernel<<<(B_ * H_ + 255) / 256, 256, 0, stream>>>(
      out_msr, msrh, B_ * H_);
  {
    int tiles = (B_ / 16) * ((H_ / 2) / 16);
    gemm_wmma_kernel<2, 1><<<tiles / 8, 256, 0, stream>>>(
        msrh, BpW1, out_b1, nullptr, hidh, B_, H_ / 2, H_);
  }
  head2_kernel<<<(B_ + 255) / 256, 256, 0, stream>>>(
      hidh, out_W2, out_b2, out);
}
